// CTCDecoder_52750788329929
// MI455X (gfx1250) — matrix operand order repaired and checked by reading, not compiled, not run on hardware
//
#include <hip/hip_runtime.h>

typedef __attribute__((ext_vector_type(16))) __bf16 v16bf;
typedef __attribute__((ext_vector_type(8)))  __bf16 v8bf;
typedef __attribute__((ext_vector_type(8)))  float  v8f;
typedef unsigned int u32x4 __attribute__((ext_vector_type(4)));
typedef int          i32x8 __attribute__((ext_vector_type(8)));
typedef int          i32x4 __attribute__((ext_vector_type(4)));

#define NEG_INF (-1.0e10f)
#define AS_STRIDE 520   // 512 + 8 bf16 pad -> row stride 260 dwords (bank spread)
#define B_ROW    520    // TDM-padded B row stride in bf16 elements
#define B_BUF_EL 8320   // 16 rows * 520 = one B buffer (16,640 B)

__device__ __forceinline__ float lae(float x, float y) {
    float m = fmaxf(x, y);
    float d = fminf(x, y) - m;
    return m + log1pf(__expf(d));
}

// ---- TDM: async DMA of one 16x512 bf16 B-tile (16 KB contiguous) into LDS,
//      with hardware padding of 4 dwords after every 256 dwords (per row). ----
__device__ __forceinline__ void tdm_load_btile(const __bf16* gsrc, unsigned ldsByteAddr) {
    unsigned long long ga = (unsigned long long)(uintptr_t)gsrc;
    // D# group0: count=1 | lds_addr | global_addr[56:0] | type=2
    u32x4 g0 = { 1u,
                 ldsByteAddr,
                 (unsigned)ga,
                 (unsigned)((ga >> 32) & 0x01FFFFFFu) | (2u << 30) };
    // D# group1: data_size=2B, pad_enable, pad_interval=7(256dw), pad_amount=3(4dw),
    //            tensor_dim0=8192, tensor_dim1=1, tile_dim0=8192, dim0_stride=8192
    i32x8 g1 = { (int)((1u << 16) | (1u << 20) | (7u << 22) | (3u << 25)),
                 (int)(8192u << 16),   // tensor_dim0[15:0] -> bits 63:48
                 (int)(1u << 16),      // tensor_dim0[31:16]=0, tensor_dim1[15:0]=1
                 (int)(8192u << 16),   // tensor_dim1[31:16]=0, tile_dim0=8192
                 0,                    // tile_dim1=0(unused), tile_dim2=0
                 8192,                 // tensor_dim0_stride[31:0]
                 0, 0 };
    i32x4 z4 = { 0, 0, 0, 0 };
    i32x8 z8 = { 0, 0, 0, 0, 0, 0, 0, 0 };
    __builtin_amdgcn_tensor_load_to_lds(g0, g1, z4, z4, z8, 0);
}

// ---------------- kernel 0: convert W (fp32 [V,H]) to bf16 ----------------
__global__ void k_cvt_w(const float* __restrict__ W, __bf16* __restrict__ Wb, int n) {
    int i = blockIdx.x * 256 + threadIdx.x;
    if (i < n) Wb[i] = (__bf16)W[i];
}

// ---------------- kernel 1: bf16 WMMA GEMM + online log-sum-exp ----------------
// grid = 2000 (M-tiles of 16 rows), single-wave blocks (32 threads).
// B tiles stream through LDS via double-buffered TDM; the A tile is staged once in LDS as bf16.
__global__ __launch_bounds__(32) void k_gemm_lse(
    const float* __restrict__ eouts, const __bf16* __restrict__ Wb,
    const float* __restrict__ bias, float* __restrict__ lse)
{
    __shared__ __align__(32) __bf16 As[16 * AS_STRIDE];   // 16,640 B
    __shared__ __align__(32) __bf16 Bs[2 * B_BUF_EL];     // 33,280 B (double buffer)

    const int lane = threadIdx.x;
    const int m0   = blockIdx.x * 16;
    const int mr   = lane & 15;
    const int hi   = lane >> 4;

    // Touch Bs with real stores (TDM writes are invisible to alias analysis).
    {
        unsigned* bz = (unsigned*)Bs;
        for (int i = lane; i < (2 * B_BUF_EL) / 2; i += 32) bz[i] = 0u;
    }

    // Stage A tile: 16 rows x 512 K, fp32 -> bf16, padded row stride.
    const float4* ev = (const float4*)(eouts + (size_t)m0 * 512);
    for (int i = lane; i < 2048; i += 32) {
        int r  = i >> 7;              // 128 float4 per row
        int c4 = i & 127;
        float4 v = ev[(size_t)r * 128 + c4];
        __bf16* dst = &As[r * AS_STRIDE + (c4 << 2)];
        dst[0] = (__bf16)v.x; dst[1] = (__bf16)v.y;
        dst[2] = (__bf16)v.z; dst[3] = (__bf16)v.w;
    }
    asm volatile("s_wait_dscnt 0x0" ::: "memory");

    const unsigned bsAddr = (unsigned)(uintptr_t)Bs;  // low 32 bits = LDS byte addr

    float mAcc[8], sAcc[8];
#pragma unroll
    for (int j = 0; j < 8; ++j) { mAcc[j] = -3.0e38f; sAcc[j] = 0.0f; }

    // Prologue: tile 0 -> buffer 0
    tdm_load_btile(Wb, bsAddr);

    for (int t = 0; t < 128; ++t) {
        if (t + 1 < 128) {
            // buffer (t+1)&1 was last read in iteration t-1; drain LDS reads, refill it
            asm volatile("s_wait_dscnt 0x0" ::: "memory");
            tdm_load_btile(Wb + (size_t)(t + 1) * (16 * 512),
                           bsAddr + (unsigned)(((t + 1) & 1) * (B_BUF_EL * 2)));
            __builtin_amdgcn_s_wait_tensorcnt(1);   // tile t has landed
        } else {
            __builtin_amdgcn_s_wait_tensorcnt(0);
        }
        asm volatile("" ::: "memory");              // order LDS reads after the wait

        // B fragment base: lane holds column (t*16 + mr); K runs [k0+hi*8, +8) and [k0+16+hi*8, +8)
        const __bf16* bp = &Bs[(t & 1) * B_BUF_EL + mr * B_ROW + hi * 8];

        v8f c = {};
#pragma unroll
        for (int kk = 0; kk < 16; ++kk) {
            const int k0 = kk * 32;
            // A fragment: row mr; K runs [k0+hi*8, +8) and [k0+16+hi*8, +8)
            const __bf16* ap = &As[mr * AS_STRIDE + k0 + hi * 8];
            v8bf alo = *(const v8bf*)ap;
            v8bf ahi = *(const v8bf*)(ap + 16);
            v16bf a = __builtin_shufflevector(alo, ahi,
                        0,1,2,3,4,5,6,7,8,9,10,11,12,13,14,15);
            v8bf blo = *(const v8bf*)(bp + k0);
            v8bf bhi = *(const v8bf*)(bp + k0 + 16);
            v16bf bm = __builtin_shufflevector(blo, bhi,
                        0,1,2,3,4,5,6,7,8,9,10,11,12,13,14,15);
            c = __builtin_amdgcn_wmma_f32_16x16x32_bf16(
                    false, a, false, bm, (short)0, c, false, false);
        }

        const float bv = bias[t * 16 + mr];
        // online max / sum-exp, 8 rows per lane (C VGPR j: rows j / j+8)
#pragma unroll
        for (int j = 0; j < 8; ++j) {
            float x  = c[j] + bv;
            float mo = mAcc[j];
            float mn = fmaxf(mo, x);
            sAcc[j]  = sAcc[j] * __expf(mo - mn) + __expf(x - mn);
            mAcc[j]  = mn;
        }
    }

    // In-wave LSE reduce: each row's 16 partials sit in one 16-lane half.
#pragma unroll
    for (int j = 0; j < 8; ++j) {
        float m = mAcc[j], s = sAcc[j];
#pragma unroll
        for (int off = 8; off >= 1; off >>= 1) {
            float m2 = __shfl_xor(m, off, 32);
            float s2 = __shfl_xor(s, off, 32);
            float mn = fmaxf(m, m2);
            s = s * __expf(m - mn) + s2 * __expf(m2 - mn);
            m = mn;
        }
        if ((lane & 15) == 0)
            lse[m0 + hi * 8 + j] = m + __logf(s);
    }
}

// ---------------- kernel 2: fp32 gathered log-probs lp_ext[b,t,s] ----------------
__global__ __launch_bounds__(256) void k_gather(
    const float* __restrict__ eouts, const float* __restrict__ W,
    const float* __restrict__ bias, const int* __restrict__ ys,
    const float* __restrict__ lse, float* __restrict__ lp)
{
    __shared__ float e[512];
    const int row = blockIdx.x;       // b*1000 + t
    const int b   = row / 1000;
    const int tid = threadIdx.x;

    e[tid]       = eouts[(size_t)row * 512 + tid];
    e[tid + 256] = eouts[(size_t)row * 512 + tid + 256];
    __syncthreads();

    const int   w     = tid >> 5;
    const int   lane  = tid & 31;
    const float l_lse = lse[row];

    for (int s = w; s < 201; s += 8) {
        const int v = (s & 1) ? ys[b * 100 + (s >> 1)] : 0;
        const float* wr = W + (size_t)v * 512;
        float acc = 0.0f;
#pragma unroll
        for (int i = 0; i < 16; ++i) {
            int k = lane + (i << 5);
            acc = fmaf(e[k], wr[k], acc);
        }
#pragma unroll
        for (int off = 16; off > 0; off >>= 1)
            acc += __shfl_xor(acc, off, 32);
        if (lane == 0)
            lp[(size_t)row * 201 + s] = acc + bias[v] - l_lse;
    }
}

// ---------------- kernel 3: CTC forward DP per batch element ----------------
__global__ __launch_bounds__(256) void k_ctc(
    const int* __restrict__ ys, const int* __restrict__ elens,
    const int* __restrict__ ylens, const float* __restrict__ lp,
    float* __restrict__ lossArr)
{
    __shared__ float a[256];
    const int b = blockIdx.x;
    const int s = threadIdx.x;
    const int elen = elens[b];
    const int ylen = ylens[b];
    const int S = 201;
    const size_t base = (size_t)b * 1000 * S;

    const bool on    = s < S;
    const bool valid = s < (2 * ylen + 1);
    bool skip = false;
    if (on && (s & 1)) {
        if (s < 3) skip = true;
        else skip = (ys[b * 100 + ((s - 1) >> 1)] != ys[b * 100 + ((s - 3) >> 1)]);
    }

    float init = NEG_INF;
    if (on && s < 2 && valid) init = lp[base + s];
    a[s] = init;
    __syncthreads();

    for (int t = 1; t < elen; ++t) {
        float nv = NEG_INF;
        if (on) {
            float x = a[s];
            if (s >= 1) x = lae(x, a[s - 1]);
            if (s >= 2 && skip) x = lae(x, a[s - 2]);
            nv = valid ? (x + lp[base + (size_t)t * S + s]) : NEG_INF;
        }
        __syncthreads();
        if (on) a[s] = nv;
        __syncthreads();
    }

    if (s == 0) {
        float last = a[2 * ylen];
        float prev = a[2 * ylen - 1];
        float loss = -lae(last, prev);
        lossArr[b] = (loss < -0.5f * NEG_INF) ? loss : 0.0f;  // zero_infinity
    }
}

// ---------------- kernel 4: deterministic final reduce ----------------
__global__ void k_final(const float* __restrict__ lossArr, float* __restrict__ out) {
    float v = lossArr[threadIdx.x];
#pragma unroll
    for (int off = 16; off > 0; off >>= 1) v += __shfl_xor(v, off, 32);
    if (threadIdx.x == 0) out[0] = v * (1.0f / 32.0f);
}

extern "C" void kernel_launch(void* const* d_in, const int* in_sizes, int n_in,
                              void* d_out, int out_size, void* d_ws, size_t ws_size,
                              hipStream_t stream) {
    const float* eouts = (const float*)d_in[0];   // [32,1000,512]
    const float* W     = (const float*)d_in[1];   // [2048,512]
    const float* bias  = (const float*)d_in[2];   // [2048]
    const int*   ys    = (const int*)d_in[3];     // [32,100]
    const int*   elens = (const int*)d_in[4];     // [32]
    const int*   ylens = (const int*)d_in[5];     // [32]
    float* out = (float*)d_out;

    // workspace layout (256B-aligned offsets), total ~27.95 MB
    char* ws = (char*)d_ws;
    __bf16* Wb      = (__bf16*)(ws + 0);                       // 2,097,152 B
    float*  lse     = (float*)(ws + 2097152);                  //   128,000 B
    float*  lp      = (float*)(ws + 2225152);                  // 25,728,000 B
    float*  lossArr = (float*)(ws + 2225152 + 25728000);       //       128 B

    k_cvt_w   <<<(2048 * 512 + 255) / 256, 256, 0, stream>>>(W, Wb, 2048 * 512);
    k_gemm_lse<<<2000, 32, 0, stream>>>(eouts, Wb, bias, lse);
    k_gather  <<<32000, 256, 0, stream>>>(eouts, W, bias, ys, lse, lp);
    k_ctc     <<<32, 256, 0, stream>>>(ys, elens, ylens, lp, lossArr);
    k_final   <<<1, 32, 0, stream>>>(lossArr, out);
}
